// Readout_81965155877098
// MI455X (gfx1250) — matrix operand order repaired and checked by reading, not compiled, not run on hardware
//
#include <hip/hip_runtime.h>
#include <hip/hip_bf16.h>

// ---------- types ----------
typedef __attribute__((ext_vector_type(16))) __bf16        bf16x16;
typedef __attribute__((ext_vector_type(16))) float         floatx16;
typedef __attribute__((ext_vector_type(8)))  float         floatx8;
typedef __attribute__((ext_vector_type(4)))  float         floatx4;
typedef __attribute__((ext_vector_type(8)))  unsigned int  uintx8;

union FragU {
    unsigned int u[8];
    uintx8       v;
    bf16x16      b;
};

#define HID   256
#define NCLS  104
#define NTILE 7          // ceil(104/16)
#define KG    512        // gate K
#define KT    256        // transform K

static __device__ __forceinline__ unsigned int pk(float lo, float hi) {
    // native fptrunc -> v_cvt_pk_bf16_f32 (RNE)
    __bf16 l = (__bf16)lo, h = (__bf16)hi;
    unsigned short ul = __builtin_bit_cast(unsigned short, l);
    unsigned short uh = __builtin_bit_cast(unsigned short, h);
    return (unsigned int)ul | ((unsigned int)uh << 16);
}

// ---------- zero the graph accumulator ----------
__global__ void zero_kernel(float* p, int n) {
    int i = blockIdx.x * blockDim.x + threadIdx.x;
    if (i < n) p[i] = 0.0f;
}

// ---------- repack f32 weights [K x 104] into bf16 WMMA B-fragments ----------
// fragment u32 index = ((kblock*NTILE + t)*32 + lane)*8 + v
// B layout used here (contiguous k, as in load_a): lane n -> N = n&15, K-half = (n>>4)*16;
//                           VGPR v holds K = half+2v (lo16), half+2v+1 (hi16)
__global__ void prep_weights_kernel(const float* __restrict__ W,
                                    unsigned int* __restrict__ frag,
                                    int K) {
    int idx = blockIdx.x * blockDim.x + threadIdx.x;
    int total = (K / 32) * NTILE * 256;
    if (idx >= total) return;
    int v      = idx & 7;
    int lane   = (idx >> 3) & 31;
    int tk     = idx >> 8;            // kblock*NTILE + t
    int t      = tk % NTILE;
    int kblock = tk / NTILE;
    int k = kblock * 32 + ((lane >> 4) * 16) + 2 * v;
    int n = t * 16 + (lane & 15);
    float lo = (n < NCLS) ? W[(size_t)k * NCLS + n]       : 0.0f;
    float hi = (n < NCLS) ? W[(size_t)(k + 1) * NCLS + n] : 0.0f;
    frag[idx] = pk(lo, hi);
}

// ---------- A fragment from f32 row-major (contiguous k order used here, same as B) ----------
// lane: M = lane&15, kbl = (lane>>4)*16; VGPRs 0-7 = K kbl+0..15
// bf16x16 element i -> VGPR i/2, half i%2 => element order is exactly
// K = kbl+0..15, so one vector fptrunc does the whole fragment.
// Loads are non-temporal: node states are streamed exactly once; keep the
// reused weight fragments / graphAcc hot in WGP$/L2 instead.
static __device__ __forceinline__ bf16x16 load_a(const float* __restrict__ row, int k0) {
    floatx4 f0 = __builtin_nontemporal_load((const floatx4*)(row + k0));
    floatx4 f1 = __builtin_nontemporal_load((const floatx4*)(row + k0 + 4));
    floatx4 f2 = __builtin_nontemporal_load((const floatx4*)(row + k0 + 8));
    floatx4 f3 = __builtin_nontemporal_load((const floatx4*)(row + k0 + 12));
    floatx16 f;
    f[0]  = f0[0]; f[1]  = f0[1]; f[2]  = f0[2]; f[3]  = f0[3];
    f[4]  = f1[0]; f[5]  = f1[1]; f[6]  = f1[2]; f[7]  = f1[3];
    f[8]  = f2[0]; f[9]  = f2[1]; f[10] = f2[2]; f[11] = f2[3];
    f[12] = f3[0]; f[13] = f3[1]; f[14] = f3[2]; f[15] = f3[3];
    return __builtin_convertvector(f, bf16x16);
}

// ---------- main: gated nodewise readout + sorted-segment scatter ----------
__global__ __launch_bounds__(128)
void node_readout_kernel(const float* __restrict__ initS,
                         const float* __restrict__ finS,
                         const int*   __restrict__ gids,
                         const unsigned int* __restrict__ bGf,
                         const unsigned int* __restrict__ bTf,
                         const float* __restrict__ bg,
                         const float* __restrict__ bt,
                         float* __restrict__ graphAcc,
                         int nTiles) {
    const int wave = threadIdx.x >> 5;
    const int lane = threadIdx.x & 31;
    const int tile = blockIdx.x * 4 + wave;
    if (tile >= nTiles) return;                 // whole-wave uniform: EXEC stays all-1s

    const int m   = lane & 15;
    const int kbl = (lane >> 4) * 16;
    const long rowBase = (long)tile * 16;
    const float* iRow = initS + (rowBase + m) * HID;
    const float* fRow = finS  + (rowBase + m) * HID;

    // fRow is first consumed 8 K-iterations from now: prefetch its 1KB row
    // (global_prefetch_b8 fetches the full cacheline per address).
#pragma unroll
    for (int off = 0; off < HID; off += 32)
        __builtin_prefetch(fRow + off, 0, 3);

    floatx8 accG[NTILE], accT[NTILE];
    const floatx8 z = {0.f, 0.f, 0.f, 0.f, 0.f, 0.f, 0.f, 0.f};
#pragma unroll
    for (int t = 0; t < NTILE; ++t) { accG[t] = z; accT[t] = z; }

    const uintx8* bG8 = (const uintx8*)bGf;
    const uintx8* bT8 = (const uintx8*)bTf;

    // gate K = 0..255 comes from initial_node_states
    for (int kk = 0; kk < 8; ++kk) {
        bf16x16 a = load_a(iRow, kk * 32 + kbl);
#pragma unroll
        for (int t = 0; t < NTILE; ++t) {
            FragU b; b.v = bG8[(kk * NTILE + t) * 32 + lane];
            accG[t] = __builtin_amdgcn_wmma_f32_16x16x32_bf16(
                false, a, false, b.b, (short)0, accG[t], false, false);
        }
    }
    // gate K = 256..511 AND transform K = 0..255 both read final_node_states once
    for (int kk = 0; kk < 8; ++kk) {
        bf16x16 a = load_a(fRow, kk * 32 + kbl);
#pragma unroll
        for (int t = 0; t < NTILE; ++t) {
            FragU b; b.v = bG8[((kk + 8) * NTILE + t) * 32 + lane];
            accG[t] = __builtin_amdgcn_wmma_f32_16x16x32_bf16(
                false, a, false, b.b, (short)0, accG[t], false, false);
        }
#pragma unroll
        for (int t = 0; t < NTILE; ++t) {
            FragU b; b.v = bT8[(kk * NTILE + t) * 32 + lane];
            accT[t] = __builtin_amdgcn_wmma_f32_16x16x32_bf16(
                false, a, false, b.b, (short)0, accT[t], false, false);
        }
    }

    // Epilogue: sigmoid gate, multiply, run-length-compacted atomic scatter.
    // C/D layout: VGPR j, lane l -> row M = (l>=16?8:0)+j, col N = l&15.
    const int mbase = (int)rowBase + (lane >> 4) * 8;
    int g8[8];
#pragma unroll
    for (int j = 0; j < 8; ++j) g8[j] = gids[mbase + j];

#pragma unroll
    for (int t = 0; t < NTILE; ++t) {
        int n = t * 16 + (lane & 15);
        if (n >= NCLS) continue;
        float bgv = bg[n], btv = bt[n];
        float pend = 0.0f;
        int cur = g8[0];
#pragma unroll
        for (int j = 0; j < 8; ++j) {
            float gl   = accG[t][j] + bgv;
            float gate = 1.0f / (1.0f + __expf(-gl));
            float val  = gate * (accT[t][j] + btv);
            if (g8[j] != cur) {                       // sorted list -> rare
                atomicAdd(graphAcc + (long)cur * NCLS + n, pend);
                pend = 0.0f; cur = g8[j];
            }
            pend += val;
        }
        atomicAdd(graphAcc + (long)cur * NCLS + n, pend);
    }
}

// ---------- per-graph: concat aux, BatchNorm (eval), MLP 106->128->104 ----------
__global__ __launch_bounds__(128)
void graph_mlp_kernel(const float* __restrict__ acc,
                      const float* __restrict__ aux,
                      const float* __restrict__ gamma, const float* __restrict__ beta,
                      const float* __restrict__ mean,  const float* __restrict__ var,
                      const float* __restrict__ W1, const float* __restrict__ b1,
                      const float* __restrict__ W2, const float* __restrict__ b2,
                      float* __restrict__ out) {
    const int g = blockIdx.x;
    const int tid = threadIdx.x;
    __shared__ float nrm[106];
    __shared__ float h[128];
    if (tid < 106) {
        float x = (tid < 104) ? acc[(size_t)g * NCLS + tid] : aux[g * 2 + (tid - 104)];
        nrm[tid] = (x - mean[tid]) * rsqrtf(var[tid] + 1e-5f) * gamma[tid] + beta[tid];
    }
    __syncthreads();
    float s = b1[tid];
#pragma unroll 2
    for (int i = 0; i < 106; ++i) s += nrm[i] * W1[i * 128 + tid];
    h[tid] = fmaxf(s, 0.0f);
    __syncthreads();
    if (tid < 104) {
        float s2 = b2[tid];
#pragma unroll 2
        for (int i = 0; i < 128; ++i) s2 += h[i] * W2[i * NCLS + tid];
        out[(size_t)g * NCLS + tid] = s2;
    }
}

// ---------- launcher ----------
extern "C" void kernel_launch(void* const* d_in, const int* in_sizes, int n_in,
                              void* d_out, int out_size, void* d_ws, size_t ws_size,
                              hipStream_t stream) {
    const float* initS = (const float*)d_in[0];
    const float* finS  = (const float*)d_in[1];
    const float* aux   = (const float*)d_in[2];
    const int*   gids  = (const int*)d_in[3];
    const float* Wg    = (const float*)d_in[5];
    const float* bg    = (const float*)d_in[6];
    const float* Wt    = (const float*)d_in[7];
    const float* bt    = (const float*)d_in[8];
    const float* bn_g  = (const float*)d_in[9];
    const float* bn_b  = (const float*)d_in[10];
    const float* bn_m  = (const float*)d_in[11];
    const float* bn_v  = (const float*)d_in[12];
    const float* W1    = (const float*)d_in[13];
    const float* b1    = (const float*)d_in[14];
    const float* W2    = (const float*)d_in[15];
    const float* b2    = (const float*)d_in[16];
    float* out = (float*)d_out;

    const int N = in_sizes[0] / HID;   // 500000
    const int G = in_sizes[2] / 2;     // 2048

    // workspace layout (all 32B-aligned)
    const size_t accBytes  = (size_t)G * NCLS * sizeof(float);            // 851,968
    const size_t fragGCnt  = (KG / 32) * NTILE * 256;                     // 28,672 u32
    const size_t fragTCnt  = (KT / 32) * NTILE * 256;                     // 14,336 u32
    char* ws = (char*)d_ws;
    float*        graphAcc = (float*)ws;
    unsigned int* fragG    = (unsigned int*)(ws + accBytes);
    unsigned int* fragT    = (unsigned int*)(ws + accBytes + fragGCnt * 4);

    const int zc = G * NCLS;
    zero_kernel<<<(zc + 255) / 256, 256, 0, stream>>>(graphAcc, zc);
    prep_weights_kernel<<<((int)fragGCnt + 255) / 256, 256, 0, stream>>>(Wg, fragG, KG);
    prep_weights_kernel<<<((int)fragTCnt + 255) / 256, 256, 0, stream>>>(Wt, fragT, KT);

    const int nTiles = (N + 15) / 16;  // 31250
    node_readout_kernel<<<(nTiles + 3) / 4, 128, 0, stream>>>(
        initS, finS, gids, fragG, fragT, bg, bt, graphAcc, nTiles);

    graph_mlp_kernel<<<G, 128, 0, stream>>>(
        graphAcc, aux, bn_g, bn_b, bn_m, bn_v, W1, b1, W2, b2, out);
}
